// DenseNet201ABNVITGAP_71562745086144
// MI455X (gfx1250) — matrix operand order repaired and checked by reading, not compiled, not run on hardware
//
#include <hip/hip_runtime.h>
#include <hip/hip_bf16.h>
#include <math.h>

typedef __attribute__((ext_vector_type(8)))  float        v8f;
typedef __attribute__((ext_vector_type(16))) __bf16       v16bf;
typedef __attribute__((ext_vector_type(4)))  unsigned int u32x4;

union FragBF {
  v16bf bf;
  u32x4 u4[2];
};

#define NTOK 49
#define CIN  896
#define PA   904                       // padded pitch (u16 elems) for A and Q in LDS
#define QK_SCALE 0.08838834764831845f  // 128^-0.5

__device__ __forceinline__ unsigned short f2bf(float f) {
  unsigned u = __float_as_uint(f);
  u += 0x7fffu + ((u >> 16) & 1u);     // round-to-nearest-even
  return (unsigned short)(u >> 16);
}

// ---------------------------------------------------------------------------
// P0: pack W_qkv rows [0,1792) transposed into bf16, in the WMMA B-fragment
// layout: Wpack[((kblk*112 + ntile)*32 + lane)*16 + j]
//   kblk = c/32, ntile = co/16, lane = (co&15) + (c%32 >= 16 ? 16 : 0), j = c%16
// q-scale folded into rows co < 896.
// ---------------------------------------------------------------------------
__global__ __launch_bounds__(256) void pack_w_kernel(const float* __restrict__ W,
                                                     unsigned short* __restrict__ Wpack) {
  int idx = blockIdx.x * 256 + threadIdx.x;   // over 1792*896
  if (idx >= 1792 * 896) return;
  int co = idx / 896;
  int c  = idx - co * 896;
  float v = W[idx] * (co < 896 ? QK_SCALE : 1.0f);
  int kb    = c >> 5;
  int c5    = c & 31;
  int ntile = co >> 4;
  int lane  = (co & 15) + (c5 & 16);
  int j     = c5 & 15;
  Wpack[(((size_t)kb * 112 + ntile) * 32 + lane) * 16 + j] = f2bf(v);
}

__global__ void zero_mask_kernel(unsigned* __restrict__ maskG) {
  if (threadIdx.x < 76) maskG[threadIdx.x] = 0u;
}

// ---------------------------------------------------------------------------
// K1: per-batch  tokens->QK projection (WMMA bf16)  +  attention scores (WMMA)
//     + softmax + min over heads  ->  fusedG[b][49*49]
// One workgroup (8 waves) per batch. W stream is software-pipelined through
// registers so the L2 fetch of chunk kb+1 overlaps the WMMAs of chunk kb.
// ---------------------------------------------------------------------------
__global__ __launch_bounds__(256) void qk_attn_kernel(const float* __restrict__ x,
                                                      const unsigned short* __restrict__ Wpack,
                                                      float* __restrict__ fusedG) {
  // LDS layout (bytes):
  //   [0,      115712) : A bf16 [64][PA]        -> later Kpack bf16 [7*4*4][32][16]
  //   [115712, 173056) : Wbuf bf16 [56*32*16]   -> later fused f32 [64*64]
  //   [173056, 288768) : Q bf16 [64][PA]
  __shared__ __align__(16) unsigned char smem[288768];
  unsigned short* A  = (unsigned short*)smem;
  unsigned short* Kp = (unsigned short*)smem;
  unsigned short* Wb = (unsigned short*)(smem + 115712);
  float*          fz = (float*)(smem + 115712);
  unsigned short* Qs = (unsigned short*)(smem + 173056);

  const int tid  = threadIdx.x;
  const int wave = tid >> 5;
  const int lane = tid & 31;
  const int g    = lane >> 4;
  const int lr   = lane & 15;
  const int b    = blockIdx.x;

  // ---- stage x[b] (contiguous 896*49 f32) -> A transposed bf16 -------------
  const float* xb = x + (size_t)b * (CIN * NTOK);
  for (int i = tid; i < CIN * NTOK; i += 256) {
    float v = xb[i];
    int c = i / NTOK;
    int n = i - c * NTOK;
    A[n * PA + c] = f2bf(v);
  }
  for (int i = tid; i < 15 * PA; i += 256) A[NTOK * PA + i] = 0;  // pad rows 49..63

  const v8f zero8 = {};

  // ---- projection GEMM: half 0 = Q cols [0,896), half 1 = K cols [896,1792)
  for (int half = 0; half < 2; ++half) {
    v8f acc[4][7];
#pragma unroll
    for (int mt = 0; mt < 4; ++mt)
#pragma unroll
      for (int nt = 0; nt < 7; ++nt) acc[mt][nt] = zero8;

    // preload chunk 0 of this half into registers (14 x 16B per thread)
    u32x4 wreg[14];
    {
      const u32x4* src = (const u32x4*)(Wpack + (size_t)(half * 56) * 512);
#pragma unroll
      for (int i = 0; i < 14; ++i) wreg[i] = src[tid + 256 * i];
    }

    for (int kb = 0; kb < 28; ++kb) {
      __syncthreads();  // previous iteration done reading Wbuf
      // deposit current chunk (56KB) into LDS
      u32x4* dst = (u32x4*)Wb;
#pragma unroll
      for (int i = 0; i < 14; ++i) dst[tid + 256 * i] = wreg[i];
      // start fetching next chunk; latency hides under the WMMAs below
      if (kb + 1 < 28) {
        const u32x4* src = (const u32x4*)(Wpack + (size_t)((kb + 1) * 112 + half * 56) * 512);
#pragma unroll
        for (int i = 0; i < 14; ++i) wreg[i] = src[tid + 256 * i];
      }
      if (kb + 2 < 28)  // warm L2/WGP$ two chunks ahead (global_prefetch_b8)
        __builtin_prefetch((const void*)(Wpack + (size_t)((kb + 2) * 112 + half * 56) * 512 + tid * 112), 0, 1);
      __syncthreads();  // Wbuf ready

      FragBF af[4];
#pragma unroll
      for (int mt = 0; mt < 4; ++mt) {
        const unsigned short* ap = A + (mt * 16 + lr) * PA + kb * 32 + 16 * g;
        af[mt].u4[0] = *(const u32x4*)(ap);        // K = 16g .. 16g+7
        af[mt].u4[1] = *(const u32x4*)(ap + 8);    // K = 16g+8 .. 16g+15
      }
#pragma unroll
      for (int nt = 0; nt < 7; ++nt) {
        FragBF bfr;
        const unsigned short* bp = Wb + ((size_t)(wave * 7 + nt) * 32 + lane) * 16;
        bfr.u4[0] = *(const u32x4*)(bp);
        bfr.u4[1] = *(const u32x4*)(bp + 8);
#pragma unroll
        for (int mt = 0; mt < 4; ++mt) {
          acc[mt][nt] = __builtin_amdgcn_wmma_f32_16x16x32_bf16(
              false, af[mt].bf, false, bfr.bf, (short)0, acc[mt][nt], false, false);
        }
      }
    }

    if (half == 0) {
      // Q epilogue: row-major bf16 [token n][feature co]
#pragma unroll
      for (int mt = 0; mt < 4; ++mt)
#pragma unroll
        for (int nt = 0; nt < 7; ++nt)
#pragma unroll
          for (int i = 0; i < 8; ++i) {
            int n  = mt * 16 + i + 8 * g;
            int co = wave * 112 + nt * 16 + lr;
            Qs[n * PA + co] = f2bf(acc[mt][nt][i]);
          }
    } else {
      __syncthreads();  // everyone done reading A/Wbuf before we alias them
      // K epilogue: scatter into attention B-fragment layout
#pragma unroll
      for (int mt = 0; mt < 4; ++mt)
#pragma unroll
        for (int nt = 0; nt < 7; ++nt)
#pragma unroll
          for (int i = 0; i < 8; ++i) {
            int m  = mt * 16 + i + 8 * g;           // token (B-frag n index)
            int dg = wave * 112 + nt * 16 + lr;     // feature 0..895
            int h  = dg >> 7;
            int dk = (dg >> 5) & 3;                 // k-block within head
            int dp = dg & 31;                       // k within block
            int idx = ((((h * 4 + dk) * 4 + mt) * 32) + ((m & 15) + (dp & 16))) * 16 + (dp & 15);
            Kp[idx] = f2bf(acc[mt][nt][i]);
          }
    }
  }

  __syncthreads();
  for (int i = tid; i < 64 * 64; i += 256) ((unsigned*)fz)[i] = 0x7F800000u;  // +inf
  __syncthreads();

  // ---- attention: wave h computes head h; S = Q_h (64x128) * K_h^T --------
  if (wave < 7) {
    const int h = wave;
    v8f sc[4][4];
#pragma unroll
    for (int mn = 0; mn < 4; ++mn)
#pragma unroll
      for (int mm = 0; mm < 4; ++mm) sc[mn][mm] = zero8;

#pragma unroll
    for (int kb = 0; kb < 4; ++kb) {
      FragBF aq[4];
#pragma unroll
      for (int mn = 0; mn < 4; ++mn) {
        const unsigned short* ap = Qs + (mn * 16 + lr) * PA + h * 128 + kb * 32 + 16 * g;
        aq[mn].u4[0] = *(const u32x4*)(ap);
        aq[mn].u4[1] = *(const u32x4*)(ap + 8);
      }
#pragma unroll
      for (int mm = 0; mm < 4; ++mm) {
        FragBF bk;
        const unsigned short* kp = Kp + ((size_t)((h * 4 + kb) * 4 + mm) * 32 + lane) * 16;
        bk.u4[0] = *(const u32x4*)(kp);
        bk.u4[1] = *(const u32x4*)(kp + 8);
#pragma unroll
        for (int mn = 0; mn < 4; ++mn)
          sc[mn][mm] = __builtin_amdgcn_wmma_f32_16x16x32_bf16(
              false, aq[mn].bf, false, bk.bf, (short)0, sc[mn][mm], false, false);
      }
    }

    // row softmax over m (49 valid) + min over heads via ds_min on float bits
    unsigned* fu = (unsigned*)fz;
#pragma unroll
    for (int mn = 0; mn < 4; ++mn) {
#pragma unroll
      for (int i = 0; i < 8; ++i) {
        int n = mn * 16 + i + 8 * g;
        float v0 = sc[mn][0][i];
        float v1 = sc[mn][1][i];
        float v2 = sc[mn][2][i];
        float v3 = (lr == 0) ? sc[mn][3][i] : -INFINITY;  // m=48 only at lr==0
        float rm = fmaxf(fmaxf(v0, v1), fmaxf(v2, v3));
#pragma unroll
        for (int d = 1; d < 16; d <<= 1) rm = fmaxf(rm, __shfl_xor(rm, d, 32));
        float e0 = __expf(v0 - rm);
        float e1 = __expf(v1 - rm);
        float e2 = __expf(v2 - rm);
        float e3 = (lr == 0) ? __expf(v3 - rm) : 0.0f;
        float rs = e0 + e1 + e2 + e3;
#pragma unroll
        for (int d = 1; d < 16; d <<= 1) rs += __shfl_xor(rs, d, 32);
        float inv = 1.0f / rs;
        if (n < NTOK) {
          atomicMin(&fu[n * 64 + lr],      __float_as_uint(e0 * inv));
          atomicMin(&fu[n * 64 + 16 + lr], __float_as_uint(e1 * inv));
          atomicMin(&fu[n * 64 + 32 + lr], __float_as_uint(e2 * inv));
          if (lr == 0) atomicMin(&fu[n * 64 + 48], __float_as_uint(e3 * inv));
        }
      }
    }
  }
  __syncthreads();

  float* gf = fusedG + (size_t)b * 2401;
  for (int i = tid; i < 2401; i += 256) {
    int n = i / 49, m = i - n * 49;
    gf[i] = fz[n * 64 + m];
  }
}

// ---------------------------------------------------------------------------
// K2: per-batch radix-select of bottom-2160 entries; OR selected (p != 0)
// positions into the shared 2401-bit mask.
// ---------------------------------------------------------------------------
__global__ __launch_bounds__(256) void select_kernel(const float* __restrict__ fusedG,
                                                     unsigned* __restrict__ maskG) {
  __shared__ unsigned vals[2401];
  __shared__ unsigned bins[256];
  __shared__ unsigned sh_prefix;
  __shared__ int sh_r, sh_less;
  const int b = blockIdx.x, tid = threadIdx.x;

  for (int i = tid; i < 2401; i += 256)
    vals[i] = __float_as_uint(fusedG[(size_t)b * 2401 + i]);  // all >= 0 -> monotone bits
  if (tid == 0) { sh_prefix = 0u; sh_r = 2160; }              // 0-based rank of threshold
  __syncthreads();

  for (int pass = 0; pass < 4; ++pass) {
    int shift = 24 - 8 * pass;
    bins[tid] = 0u;
    __syncthreads();
    unsigned pmask = (pass == 0) ? 0u : (0xFFFFFFFFu << (shift + 8));
    unsigned pref  = sh_prefix;
    for (int i = tid; i < 2401; i += 256) {
      unsigned v = vals[i];
      if ((v & pmask) == pref) atomicAdd(&bins[(v >> shift) & 255], 1u);
    }
    __syncthreads();
    if (tid == 0) {
      int r = sh_r;
      int v = 0;
      for (; v < 255; ++v) {
        if (r < (int)bins[v]) break;
        r -= (int)bins[v];
      }
      sh_prefix = pref | ((unsigned)v << shift);
      sh_r = r;
    }
    __syncthreads();
  }
  const unsigned T = sh_prefix;  // bit pattern of the rank-2160 value

  if (tid == 0) sh_less = 0;
  __syncthreads();
  {
    int lc = 0;
    for (int i = tid; i < 2401; i += 256) lc += (vals[i] < T) ? 1 : 0;
    atomicAdd(&sh_less, lc);
  }
  __syncthreads();
  const int extra = 2160 - sh_less;  // ties at T selected in ascending index order

  for (int p = tid; p < 2401; p += 256) {
    unsigned v = vals[p];
    bool sel = (v < T);
    if (v == T) {
      int ord = 0;
      for (int q = 0; q < p; ++q) ord += (vals[q] == T) ? 1 : 0;
      sel = (ord < extra);
    }
    if (sel && p != 0) atomicOr(&maskG[p >> 5], 1u << (p & 31));
  }
}

// ---------------------------------------------------------------------------
// K3: masking (batch 0 only) + rollout normalization -> att[b][49]
//   a = (fused + I)/2 ; s[i] = row-sum ; att[j] = colsum[j] / (49 * s[j])
// ---------------------------------------------------------------------------
__global__ __launch_bounds__(256) void finalize_kernel(const float* __restrict__ fusedG,
                                                       const unsigned* __restrict__ maskG,
                                                       float* __restrict__ attG) {
  __shared__ float fv[2401];
  __shared__ float s[49];
  const int b = blockIdx.x, tid = threadIdx.x;
  for (int i = tid; i < 2401; i += 256) {
    float v = fusedG[(size_t)b * 2401 + i];
    if (b == 0 && ((maskG[i >> 5] >> (i & 31)) & 1u)) v = 0.0f;
    int r = i / 49, c = i - r * 49;
    fv[i] = 0.5f * (v + (r == c ? 1.0f : 0.0f));
  }
  __syncthreads();
  if (tid < 49) {
    float ss = 0.0f;
    for (int j = 0; j < 49; ++j) ss += fv[tid * 49 + j];
    s[tid] = ss;
  }
  __syncthreads();
  if (tid < 49) {
    float cs = 0.0f;
    for (int i = 0; i < 49; ++i) cs += fv[i * 49 + tid];
    attG[(size_t)b * 49 + tid] = cs / (49.0f * s[tid]);
  }
}

// ---------------------------------------------------------------------------
// K4: rx = x * (1 + att[b][n]), float4-vectorized (exact grid: 43904 x 256)
// ---------------------------------------------------------------------------
__global__ __launch_bounds__(256) void out_kernel(const float* __restrict__ x,
                                                  const float* __restrict__ attG,
                                                  float* __restrict__ out) {
  size_t i4 = (size_t)blockIdx.x * 256 + threadIdx.x;  // 11,239,424 float4 total
  float4 v = ((const float4*)x)[i4];
  size_t base = i4 * 4;
  size_t b = base / 43904;                 // batch constant within a float4
  int r  = (int)(base - b * 43904);
  int n  = r % 49;
  const float* att = attG + b * 49;
  float4 o;
  o.x = v.x * (1.0f + att[n]); n = (n + 1 == 49) ? 0 : n + 1;
  o.y = v.y * (1.0f + att[n]); n = (n + 1 == 49) ? 0 : n + 1;
  o.z = v.z * (1.0f + att[n]); n = (n + 1 == 49) ? 0 : n + 1;
  o.w = v.w * (1.0f + att[n]);
  ((float4*)out)[i4] = o;
}

// ---------------------------------------------------------------------------
extern "C" void kernel_launch(void* const* d_in, const int* in_sizes, int n_in,
                              void* d_out, int out_size, void* d_ws, size_t ws_size,
                              hipStream_t stream) {
  (void)in_sizes; (void)n_in; (void)out_size; (void)ws_size;
  const float* x = (const float*)d_in[0];       // [1024,896,7,7]
  const float* W = (const float*)d_in[1];       // [2688,896]
  float* out = (float*)d_out;                   // [1024,896,7,7]

  // workspace layout
  unsigned short* Wpack = (unsigned short*)d_ws;                       // 3,211,264 B
  float*    fusedG = (float*)((char*)d_ws + 3211264);                  // 9,834,496 B
  unsigned* maskG  = (unsigned*)((char*)d_ws + 13045760);              // 512 B
  float*    attG   = (float*)((char*)d_ws + 13046272);                 // 200,704 B

  pack_w_kernel<<<dim3(6272), dim3(256), 0, stream>>>(W, Wpack);
  zero_mask_kernel<<<dim3(1), dim3(256), 0, stream>>>(maskG);
  qk_attn_kernel<<<dim3(1024), dim3(256), 0, stream>>>(x, Wpack, fusedG);
  select_kernel<<<dim3(1024), dim3(256), 0, stream>>>(fusedG, maskG);
  finalize_kernel<<<dim3(1024), dim3(256), 0, stream>>>(fusedG, maskG, attG);
  out_kernel<<<dim3(43904), dim3(256), 0, stream>>>(x, attG, out);
}
